// NetGINE_13340168421427
// MI455X (gfx1250) — matrix operand order repaired and checked by reading, not compiled, not run on hardware
//
#include <hip/hip_runtime.h>
#include <hip/hip_bf16.h>
#include <math.h>

typedef __attribute__((ext_vector_type(16))) _Float16 v16h;
typedef __attribute__((ext_vector_type(8)))  float    v8f;

#define DIMP 64   // padded feature dim (all layers)
#define WAVES 8   // waves per block (256 threads, wave32)

// ---------------- WMMA helpers (CDNA5 16x16x32 f16 -> f32) ----------------
__device__ __forceinline__ v8f wmma16(v16h a, v16h b, v8f c) {
  // (neg_a, A, neg_b, B, c_mod, C, reuse_a, reuse_b)
  return __builtin_amdgcn_wmma_f32_16x16x32_f16(false, a, false, b, (short)0, c, false, false);
}
// A fragment element -> K index (16-bit A 16x32, ISA 7.12.2)
__device__ __forceinline__ int kofA(int j, int e, int half) {
  return ((j < 4) ? 0 : 16) + 8 * half + 2 * (j & 3) + e;
}
// A-fragment-ordered address for matrix element (row m, k):
//   fragment kt = k>>5; owner lane = ((k>>3)&1)*16 + m; elem = ((k&16)>>1)|(k&7)
__device__ __forceinline__ int faddr(int m, int k) {
  int kt = k >> 5, kk = k & 31;
  int half = (kk >> 3) & 1;
  int eidx = ((kk & 16) >> 1) | (kk & 7);
  return ((kt * 32 + half * 16 + m) << 4) + eidx;
}

// ---------------- weight packing ---------------------------------------------
// W[dout,din] (f32) -> W^T padded 64x64 f16, stored in B-fragment-major order:
//   dst[(((kt*4+tt)*32+lane)<<4) + elem]  where k = 32*kt + 16*(elem>>3) + 8*(lane>>4) + (elem&7),
//   n = tt*16 + (lane&15).  Each lane's 16 halves are contiguous (one v16h).
__global__ void pack_wT_kernel(const float* __restrict__ W, _Float16* __restrict__ dst,
                               int din, int dout) {
  int i = blockIdx.x * 256 + threadIdx.x;
  if (i < DIMP * DIMP) {
    int elem = i & 15;
    int lane = (i >> 4) & 31;
    int ft   = (i >> 9) & 7;          // kt*4 + tt
    int kt = ft >> 2, tt = ft & 3;
    int k = 32 * kt + 16 * (elem >> 3) + 8 * (lane >> 4) + (elem & 7);
    int n = tt * 16 + (lane & 15);
    float v = (k < din && n < dout) ? W[n * din + k] : 0.0f;
    dst[i] = (_Float16)v;
  }
}
__global__ void pack_bias_kernel(const float* __restrict__ b, float* __restrict__ dst, int dout) {
  int i = threadIdx.x;
  if (i < DIMP) dst[i] = (i < dout) ? b[i] : 0.0f;
}
__global__ void pad_x_kernel(const float* __restrict__ x, float* __restrict__ xp,
                             int Nn, int F) {
  long i = (long)blockIdx.x * 256 + threadIdx.x;
  if (i < (long)Nn * DIMP) {
    int col = (int)(i & 63);
    long row = i >> 6;
    xp[i] = (col < F) ? x[row * F + col] : 0.0f;
  }
}
__global__ void zero_f32_kernel(float* __restrict__ p, long n) {
  long i = (long)blockIdx.x * 256 + threadIdx.x;
  if (i < n) p[i] = 0.0f;
}

// ---------------- fused edge kernel -----------------------------------------
// per 16-edge tile (one wave):
//   H1 = relu(pad(edge_attr) @ W1p + b1)   (WMMA, K-tile 0 only: k<4)
//   EE = H1 @ W2p + b2                     (WMMA, 2 K-tiles)
//   msg = relu(x[src] + EE);  atomicAdd(agg[dst], msg)
__global__ __launch_bounds__(256) void edge_conv_kernel(
    const float* __restrict__ xin, const int* __restrict__ srcI, const int* __restrict__ dstI,
    const float* __restrict__ ea,
    const _Float16* __restrict__ W1p, const float* __restrict__ b1,
    const _Float16* __restrict__ W2p, const float* __restrict__ b2,
    float* __restrict__ agg, int E)
{
  __shared__ v16h sW1v[256];               // B-fragment-major weight panel 1
  __shared__ v16h sW2v[256];               // B-fragment-major weight panel 2
  __shared__ float sb1[DIMP], sb2[DIMP];
  __shared__ v16h sHv[WAVES][64];          // per-wave activation tile, A-fragment order
  __shared__ int  sSD[WAVES][32];          // [0:16) = src, [16:32) = dst

  const int t = threadIdx.x;
  const int wave = t >> 5, lane = t & 31, half = lane >> 4, mrow = lane & 15;
  const long e0 = ((long)blockIdx.x * WAVES + wave) * 16;

  // vector (B128) cooperative copy of pre-swizzled weight panels into LDS
  {
    const uint4* gw1 = (const uint4*)W1p;
    const uint4* gw2 = (const uint4*)W2p;
    uint4* lw1 = (uint4*)sW1v;
    uint4* lw2 = (uint4*)sW2v;
    for (int i = t; i < 512; i += 256) { lw1[i] = gw1[i]; lw2[i] = gw2[i]; }
  }
  if (t < DIMP) { sb1[t] = b1[t]; sb2[t] = b2[t]; }
  // stage src/dst indices for this wave's 16 edges (lanes 0-15: src, 16-31: dst)
  // and prefetch the gather/scatter rows so the WMMA work hides their latency:
  //   lanes 0-15  -> x[src] row (read path),  lanes 16-31 -> agg[dst] row (L2 RMW)
  {
    long me = e0 + mrow;
    int v = 0;
    if (me < (long)E) {
      if (half == 0) {
        v = srcI[me];
        const float* p = xin + (long)v * DIMP;
        __builtin_prefetch(p, 0, 3);          // global_prefetch_b8 (first 128B)
        __builtin_prefetch(p + 32, 0, 3);     // second 128B of the row
      } else {
        v = dstI[me];
        const float* p = agg + (long)v * DIMP;
        __builtin_prefetch(p, 1, 3);
        __builtin_prefetch(p + 32, 1, 3);
      }
    }
    sSD[wave][lane] = v;
  }
  __syncthreads();

  // A fragment from edge attrs (only k<4 nonzero; K-tile 0 suffices)
  v16h Aa;
  {
    const long me = e0 + mrow;
    const bool ok = me < (long)E;
#pragma unroll
    for (int j = 0; j < 8; ++j)
#pragma unroll
      for (int e = 0; e < 2; ++e) {
        int k = kofA(j, e, half);
        float f = (ok && k < 4) ? ea[me * 4 + k] : 0.0f;
        Aa[2 * j + e] = (_Float16)f;
      }
  }
  // GEMM1: 4 N-tiles (kt=0 only), bias splat into C
  v8f acc[4];
#pragma unroll
  for (int tt = 0; tt < 4; ++tt) {
    const int n = tt * 16 + mrow;
    v8f c;
#pragma unroll
    for (int r = 0; r < 8; ++r) c[r] = sb1[n];
    v16h Bf = sW1v[tt * 32 + lane];        // one v16h LDS vector load
    acc[tt] = wmma16(Aa, Bf, c);
  }
  // relu -> LDS, scattered directly into A-fragment order
  _Float16* sHh = (_Float16*)&sHv[wave][0];
#pragma unroll
  for (int tt = 0; tt < 4; ++tt)
#pragma unroll
    for (int r = 0; r < 8; ++r) {
      int m = r + 8 * half, k = tt * 16 + mrow;
      float f = acc[tt][r];
      sHh[faddr(m, k)] = (_Float16)(f > 0.0f ? f : 0.0f);
    }
  __syncthreads();

  v16h A2[2];
#pragma unroll
  for (int kt = 0; kt < 2; ++kt) A2[kt] = sHv[wave][kt * 32 + lane];

  v8f acc2[4];
#pragma unroll
  for (int tt = 0; tt < 4; ++tt) {
    const int n = tt * 16 + mrow;
    v8f c;
#pragma unroll
    for (int r = 0; r < 8; ++r) c[r] = sb2[n];
#pragma unroll
    for (int kt = 0; kt < 2; ++kt) {
      v16h Bf = sW2v[(kt * 4 + tt) * 32 + lane];
      c = wmma16(A2[kt], Bf, c);
    }
    acc2[tt] = c;
  }
  // gather x[src], relu, scatter-add to agg[dst]
#pragma unroll
  for (int r = 0; r < 8; ++r) {
    int m = r + 8 * half;
    long edge = e0 + m;
    if (edge < (long)E) {
      long s = sSD[wave][m];
      long d = sSD[wave][16 + m];
#pragma unroll
      for (int tt = 0; tt < 4; ++tt) {
        int n = tt * 16 + mrow;
        float v = acc2[tt][r] + xin[s * DIMP + n];
        v = v > 0.0f ? v : 0.0f;
        atomicAdd(&agg[d * DIMP + n], v);
      }
    }
  }
}

// ---------------- fused node kernel -----------------------------------------
// h = (1+eps)*x + agg ; out = relu( relu(h@M1p+c1) @ M2p + c2 )
__global__ __launch_bounds__(256) void node_mlp_kernel(
    const float* __restrict__ xin, const float* __restrict__ agg,
    const float* __restrict__ epsp,
    const _Float16* __restrict__ M1p, const float* __restrict__ c1,
    const _Float16* __restrict__ M2p, const float* __restrict__ c2,
    float* __restrict__ xout, int Nn)
{
  __shared__ v16h sW1v[256];
  __shared__ v16h sW2v[256];
  __shared__ float sb1[DIMP], sb2[DIMP];
  __shared__ v16h sHv[WAVES][64];

  const int t = threadIdx.x;
  {
    const uint4* gw1 = (const uint4*)M1p;
    const uint4* gw2 = (const uint4*)M2p;
    uint4* lw1 = (uint4*)sW1v;
    uint4* lw2 = (uint4*)sW2v;
    for (int i = t; i < 512; i += 256) { lw1[i] = gw1[i]; lw2[i] = gw2[i]; }
  }
  if (t < DIMP) { sb1[t] = c1[t]; sb2[t] = c2[t]; }
  const float epv = 1.0f + epsp[0];

  const int wave = t >> 5, lane = t & 31, half = lane >> 4, mrow = lane & 15;
  const long n0 = ((long)blockIdx.x * WAVES + wave) * 16;
  _Float16* sHh = (_Float16*)&sHv[wave][0];

  // stage h tile (f32 -> f16), scattered directly into A-fragment order
  for (int i = lane; i < 16 * DIMP; i += 32) {
    long gi = n0 + (i >> 6);
    int m = i >> 6, k = i & 63;
    float v = 0.0f;
    if (gi < (long)Nn) v = epv * xin[gi * DIMP + k] + agg[gi * DIMP + k];
    sHh[faddr(m, k)] = (_Float16)v;
  }
  __syncthreads();

  v16h A[2];
#pragma unroll
  for (int kt = 0; kt < 2; ++kt) A[kt] = sHv[wave][kt * 32 + lane];

  v8f acc[4];
#pragma unroll
  for (int tt = 0; tt < 4; ++tt) {
    const int n = tt * 16 + mrow;
    v8f c;
#pragma unroll
    for (int r = 0; r < 8; ++r) c[r] = sb1[n];
#pragma unroll
    for (int kt = 0; kt < 2; ++kt) {
      v16h Bf = sW1v[(kt * 4 + tt) * 32 + lane];
      c = wmma16(A[kt], Bf, c);
    }
    acc[tt] = c;
  }
  __syncthreads();
#pragma unroll
  for (int tt = 0; tt < 4; ++tt)
#pragma unroll
    for (int r = 0; r < 8; ++r) {
      int m = r + 8 * half, k = tt * 16 + mrow;
      float f = acc[tt][r];
      sHh[faddr(m, k)] = (_Float16)(f > 0.0f ? f : 0.0f);
    }
  __syncthreads();

  v16h A2[2];
#pragma unroll
  for (int kt = 0; kt < 2; ++kt) A2[kt] = sHv[wave][kt * 32 + lane];

#pragma unroll
  for (int tt = 0; tt < 4; ++tt) {
    const int n = tt * 16 + mrow;
    v8f c;
#pragma unroll
    for (int r = 0; r < 8; ++r) c[r] = sb2[n];
#pragma unroll
    for (int kt = 0; kt < 2; ++kt) {
      v16h Bf = sW2v[(kt * 4 + tt) * 32 + lane];
      c = wmma16(A2[kt], Bf, c);
    }
#pragma unroll
    for (int r = 0; r < 8; ++r) {
      long gi = n0 + r + 8 * half;
      if (gi < (long)Nn) {
        float v = c[r];
        xout[gi * DIMP + n] = v > 0.0f ? v : 0.0f;
      }
    }
  }
}

// ---------------- Set2Set ----------------------------------------------------
__device__ __forceinline__ unsigned fenc(float f) {
  unsigned u = __float_as_uint(f);
  return (u & 0x80000000u) ? ~u : (u | 0x80000000u);
}
__device__ __forceinline__ float fdec(unsigned k) {
  return __uint_as_float((k & 0x80000000u) ? (k ^ 0x80000000u) : ~k);
}

__global__ void lstm_kernel(const float* __restrict__ qstar, float* __restrict__ h,
                            float* __restrict__ c,
                            const float* __restrict__ Wih, const float* __restrict__ Whh,
                            const float* __restrict__ bih, const float* __restrict__ bhh)
{
  __shared__ float sq[128], sh[64], sg[256];
  const int b = blockIdx.x, t = threadIdx.x;
  if (t < 128) sq[t] = qstar[(long)b * 128 + t];
  if (t < 64)  sh[t] = h[(long)b * 64 + t];
  __syncthreads();
  float acc = bih[t] + bhh[t];
  for (int k = 0; k < 128; ++k) acc += Wih[t * 128 + k] * sq[k];
  for (int k = 0; k < 64; ++k)  acc += Whh[t * 64 + k] * sh[k];
  sg[t] = acc;
  __syncthreads();
  if (t < 64) {
    float gi = 1.0f / (1.0f + __expf(-sg[t]));
    float gf = 1.0f / (1.0f + __expf(-sg[64 + t]));
    float gg = tanhf(sg[128 + t]);
    float go = 1.0f / (1.0f + __expf(-sg[192 + t]));
    float cn = gf * c[(long)b * 64 + t] + gi * gg;
    c[(long)b * 64 + t] = cn;
    h[(long)b * 64 + t] = go * tanhf(cn);
  }
}

__global__ void s2s_step_init(float* __restrict__ r, float* __restrict__ denom,
                              unsigned* __restrict__ emax, int Bn)
{
  int i = blockIdx.x * 256 + threadIdx.x;
  if (i < Bn * 64) r[i] = 0.0f;
  if (i < Bn) { denom[i] = 0.0f; emax[i] = 0x007FFFFFu; /* enc(-inf) */ }
}

__global__ void attn_e_kernel(const float* __restrict__ x, const float* __restrict__ h,
                              const int* __restrict__ batch, float* __restrict__ ev,
                              unsigned* __restrict__ emax, int Nn)
{
  long i = (long)blockIdx.x * 256 + threadIdx.x;
  if (i >= (long)Nn) return;
  int b = batch[i];
  const float* xr = x + i * DIMP;
  const float* hr = h + (long)b * 64;
  float s = 0.0f;
#pragma unroll
  for (int c = 0; c < 64; ++c) s += xr[c] * hr[c];
  ev[i] = s;
  atomicMax(&emax[b], fenc(s));
}

__global__ void attn_a_kernel(const float* __restrict__ ev, const int* __restrict__ batch,
                              const unsigned* __restrict__ emax, float* __restrict__ av,
                              float* __restrict__ denom, int Nn)
{
  long i = (long)blockIdx.x * 256 + threadIdx.x;
  if (i >= (long)Nn) return;
  int b = batch[i];
  float v = __expf(ev[i] - fdec(emax[b]));
  av[i] = v;
  atomicAdd(&denom[b], v);
}

__global__ void attn_r_kernel(const float* __restrict__ x, const float* __restrict__ av,
                              const float* __restrict__ denom, const int* __restrict__ batch,
                              float* __restrict__ r, int Nn)
{
  long idx = (long)blockIdx.x * 256 + threadIdx.x;
  if (idx >= (long)Nn * 64) return;
  long i = idx >> 6;
  int c = (int)(idx & 63);
  int b = batch[i];
  float w = av[i] / denom[b];
  atomicAdd(&r[(long)b * 64 + c], w * x[i * DIMP + c]);
}

__global__ void qstar_kernel(const float* __restrict__ h, const float* __restrict__ r,
                             float* __restrict__ qstar, int Bn)
{
  int idx = blockIdx.x * 256 + threadIdx.x;
  if (idx >= Bn * 128) return;
  int b = idx >> 7, k = idx & 127;
  qstar[idx] = (k < 64) ? h[b * 64 + k] : r[b * 64 + (k - 64)];
}

// ---------------- readout ----------------------------------------------------
__global__ void readout_kernel(const float* __restrict__ qstar,
                               const float* __restrict__ fc1W, const float* __restrict__ fc1b,
                               const float* __restrict__ fc4W, const float* __restrict__ fc4b,
                               float* __restrict__ out)
{
  __shared__ float sq[128], st[64];
  const int b = blockIdx.x, t = threadIdx.x;
  sq[t] = qstar[(long)b * 128 + t];
  __syncthreads();
  if (t < 64) {
    float acc = fc1b[t];
    for (int k = 0; k < 128; ++k) acc += fc1W[t * 128 + k] * sq[k];
    st[t] = acc > 0.0f ? acc : 0.0f;
  }
  __syncthreads();
  if (t < 12) {
    float acc = fc4b[t];
    for (int k = 0; k < 64; ++k) acc += fc4W[t * 64 + k] * st[k];
    out[(long)b * 12 + t] = acc;
  }
}

// ---------------- host orchestration ----------------------------------------
extern "C" void kernel_launch(void* const* d_in, const int* in_sizes, int n_in,
                              void* d_out, int out_size, void* d_ws, size_t ws_size,
                              hipStream_t stream)
{
  const float* x      = (const float*)d_in[0];
  const int*   ei     = (const int*)d_in[1];
  const float* ea     = (const float*)d_in[2];
  const int*   batch  = (const int*)d_in[3];
  const int N  = in_sizes[0] / 6;
  const int E  = in_sizes[1] / 2;
  const int Bn = out_size / 12;
  const int* srcI = ei;
  const int* dstI = ei + E;

  // params: insertion-order flattening of the nested dict
  int pi = 4;
  const float *cW[6][4], *cb[6][4], *ceps[6];
  for (int l = 0; l < 6; ++l) {
    cW[l][0] = (const float*)d_in[pi + 0]; cb[l][0] = (const float*)d_in[pi + 1]; // be1
    cW[l][1] = (const float*)d_in[pi + 2]; cb[l][1] = (const float*)d_in[pi + 3]; // be2
    cW[l][2] = (const float*)d_in[pi + 4]; cb[l][2] = (const float*)d_in[pi + 5]; // mlp1
    cW[l][3] = (const float*)d_in[pi + 6]; cb[l][3] = (const float*)d_in[pi + 7]; // mlp2
    ceps[l]  = (const float*)d_in[pi + 8];
    pi += 9;
  }
  const float* Wih  = (const float*)d_in[pi++];
  const float* Whh  = (const float*)d_in[pi++];
  const float* bih  = (const float*)d_in[pi++];
  const float* bhh  = (const float*)d_in[pi++];
  const float* fc1W = (const float*)d_in[pi++];
  const float* fc1b = (const float*)d_in[pi++];
  const float* fc4W = (const float*)d_in[pi++];
  const float* fc4b = (const float*)d_in[pi++];

  // workspace bump allocator (256B aligned)
  char* wp = (char*)d_ws;
  auto walloc = [&](size_t bytes) -> void* {
    void* r = (void*)wp;
    wp += ((bytes + 255) / 256) * 256;
    return r;
  };
  float*     xA    = (float*)walloc((size_t)N * DIMP * 4);
  float*     xB    = (float*)walloc((size_t)N * DIMP * 4);
  float*     agg   = (float*)walloc((size_t)N * DIMP * 4);
  _Float16*  Wpack = (_Float16*)walloc((size_t)6 * 4 * DIMP * DIMP * 2);
  float*     Bpack = (float*)walloc((size_t)6 * 4 * DIMP * 4);
  float*     s2h   = (float*)walloc((size_t)Bn * 64 * 4);
  float*     s2c   = (float*)walloc((size_t)Bn * 64 * 4);
  float*     qstar = (float*)walloc((size_t)Bn * 128 * 4);
  float*     rbuf  = (float*)walloc((size_t)Bn * 64 * 4);
  float*     denom = (float*)walloc((size_t)Bn * 4);
  unsigned*  emax  = (unsigned*)walloc((size_t)Bn * 4);
  float*     evec  = (float*)walloc((size_t)N * 4);
  float*     avec  = (float*)walloc((size_t)N * 4);
  (void)ws_size; (void)n_in;

  // ---- pack weights into B-fragment-major f16 panels ----
  for (int l = 0; l < 6; ++l) {
    const int d1 = (l == 0) ? 6 : 64;
    const int din[4]  = { 4, d1, d1, d1 };
    const int dout[4] = { d1, d1, d1, 64 };
    for (int m = 0; m < 4; ++m) {
      _Float16* dst = Wpack + ((size_t)(l * 4 + m)) * DIMP * DIMP;
      float*    bd  = Bpack + ((size_t)(l * 4 + m)) * DIMP;
      pack_wT_kernel<<<(DIMP * DIMP + 255) / 256, 256, 0, stream>>>(cW[l][m], dst, din[m], dout[m]);
      pack_bias_kernel<<<1, 64, 0, stream>>>(cb[l][m], bd, dout[m]);
    }
  }
  // pad input features into xA [N,64]
  pad_x_kernel<<<(int)(((long)N * DIMP + 255) / 256), 256, 0, stream>>>(x, xA, N, 6);

  // ---- 6 GIN conv layers ----
  float* xCur = xA;
  float* xNxt = xB;
  const int edgeBlocks = (E + 16 * WAVES - 1) / (16 * WAVES);
  const int nodeBlocks = (N + 16 * WAVES - 1) / (16 * WAVES);
  for (int l = 0; l < 6; ++l) {
    const _Float16* W1p = Wpack + ((size_t)(l * 4 + 0)) * DIMP * DIMP;
    const _Float16* W2p = Wpack + ((size_t)(l * 4 + 1)) * DIMP * DIMP;
    const _Float16* M1p = Wpack + ((size_t)(l * 4 + 2)) * DIMP * DIMP;
    const _Float16* M2p = Wpack + ((size_t)(l * 4 + 3)) * DIMP * DIMP;
    const float* b1 = Bpack + (size_t)(l * 4 + 0) * DIMP;
    const float* b2 = Bpack + (size_t)(l * 4 + 1) * DIMP;
    const float* c1 = Bpack + (size_t)(l * 4 + 2) * DIMP;
    const float* c2 = Bpack + (size_t)(l * 4 + 3) * DIMP;

    zero_f32_kernel<<<(int)(((long)N * DIMP + 255) / 256), 256, 0, stream>>>(agg, (long)N * DIMP);
    edge_conv_kernel<<<edgeBlocks, 256, 0, stream>>>(xCur, srcI, dstI, ea,
                                                     W1p, b1, W2p, b2, agg, E);
    node_mlp_kernel<<<nodeBlocks, 256, 0, stream>>>(xCur, agg, ceps[l],
                                                    M1p, c1, M2p, c2, xNxt, N);
    float* tmp = xCur; xCur = xNxt; xNxt = tmp;
  }

  // ---- Set2Set (6 steps) ----
  zero_f32_kernel<<<(Bn * 64 + 255) / 256, 256, 0, stream>>>(s2h, (long)Bn * 64);
  zero_f32_kernel<<<(Bn * 64 + 255) / 256, 256, 0, stream>>>(s2c, (long)Bn * 64);
  zero_f32_kernel<<<(Bn * 128 + 255) / 256, 256, 0, stream>>>(qstar, (long)Bn * 128);
  const int nBlk  = (N + 255) / 256;
  const int ncBlk = (int)(((long)N * 64 + 255) / 256);
  for (int s = 0; s < 6; ++s) {
    lstm_kernel<<<Bn, 256, 0, stream>>>(qstar, s2h, s2c, Wih, Whh, bih, bhh);
    s2s_step_init<<<(Bn * 64 + 255) / 256, 256, 0, stream>>>(rbuf, denom, emax, Bn);
    attn_e_kernel<<<nBlk, 256, 0, stream>>>(xCur, s2h, batch, evec, emax, N);
    attn_a_kernel<<<nBlk, 256, 0, stream>>>(evec, batch, emax, avec, denom, N);
    attn_r_kernel<<<ncBlk, 256, 0, stream>>>(xCur, avec, denom, batch, rbuf, N);
    qstar_kernel<<<(Bn * 128 + 255) / 256, 256, 0, stream>>>(s2h, rbuf, qstar, Bn);
  }

  // ---- readout ----
  readout_kernel<<<Bn, 128, 0, stream>>>(qstar, fc1W, fc1b, fc4W, fc4b, (float*)d_out);
}
